// StgcnLayer_18820546691089
// MI455X (gfx1250) — matrix operand order repaired and checked by reading, not compiled, not run on hardware
//
#include <hip/hip_runtime.h>
#include <hip/hip_bf16.h>

typedef __attribute__((ext_vector_type(16))) __bf16 v16bf;
typedef __attribute__((ext_vector_type(8)))  __bf16 v8bf;
typedef __attribute__((ext_vector_type(8)))  float  v8f;

#define NN     32
#define CI     64
#define LTIME  300
#define VJ     25
#define CO     64
#define PP     3
#define KSTEPS 50              // K = 1600 = 50 * 32
#define COLB   7               // column blocks of 256 (covers 1792, pad zeroed)
#define CTILES (COLB * 16)     // 112 16-wide column tiles
#define NCOL   1600
#define LTILES 19              // ceil(300 / 16) row tiles per n
#define LTPAD  21              // padded to row-groups of 3 (7 groups)
#define LROWS  304             // LTILES * 16
#define EPSBN  1e-3f

#define MC_ELEMS (KSTEPS * CTILES * 32 * 16)       // 2,867,200 bf16
#define XF_ELEMS (NN * LTPAD * KSTEPS * 32 * 16)   // 17,203,200 bf16
#define MC_BYTES (MC_ELEMS * 2)                    // 5.5 MB (multiple of 256)
#define XF_BYTES (XF_ELEMS * 2)                    // 32.8 MB (multiple of 256)

// ---------------------------------------------------------------------------
// Kernel 1: combined matrix M[(cin,v),(c,w)] = sum_p W[cin,c*3+p]*A[p,v,w],
// stored in WMMA B-fragment lane order (bf16). Cols >= 1600 zero padded.
//   lane<16: K = kstep*32 + vidx ; lane>=16: K = kstep*32 + 16 + vidx
// ---------------------------------------------------------------------------
__global__ __launch_bounds__(256) void build_mc_kernel(
    const float* __restrict__ A, const float* __restrict__ W,
    __bf16* __restrict__ mc) {
  int idx = blockIdx.x * 256 + threadIdx.x;
  if (idx >= MC_ELEMS) return;
  int vidx = idx & 15;
  int rest = idx >> 4;
  int L    = rest & 31;  rest >>= 5;
  int t    = rest % CTILES;
  int k    = rest / CTILES;

  int col = t * 16 + (L & 15);
  int kk  = k * 32 + ((L >> 4) << 4) + vidx;

  float val = 0.f;
  if (col < NCOL) {
    int cin = kk / VJ, v = kk - cin * VJ;
    int c   = col / VJ, w = col - c * VJ;
#pragma unroll
    for (int p = 0; p < PP; ++p)
      val += W[cin * (CO * PP) + c * PP + p] * A[(p * VJ + v) * VJ + w];
  }
  mc[idx] = (__bf16)val;
}

// ---------------------------------------------------------------------------
// Kernel 2: convert x into A-fragment-ordered bf16 tiles (16-row l-tiles,
// no halo). Layout: v16bf index = ((n*LTPAD + lt)*KSTEPS + kb)*32 + lane.
//   lane<16: K = vidx ; lane>=16: K = 16 + vidx
// Rows l >= 300 (incl. pad tiles 19,20) are zero.
// Each thread gathers 8 consecutive kk (memory-contiguous modulo one v-wrap)
// and writes one 16-byte store.
// ---------------------------------------------------------------------------
__global__ __launch_bounds__(256) void build_xfrag_kernel(
    const float* __restrict__ x, __bf16* __restrict__ xf) {
  int t = blockIdx.x * 256 + threadIdx.x;
  const int total = NN * LTPAD * KSTEPS * 32 * 2;
  if (t >= total) return;
  int half = t & 1;
  int rest = t >> 1;
  int L  = rest & 31;   rest >>= 5;
  int kb = rest % KSTEPS; rest /= KSTEPS;
  int lt = rest % LTPAD;
  int n  = rest / LTPAD;

  int l   = lt * 16 + (L & 15);
  int Kb  = ((L < 16) ? 0 : 16) + half * 8;
  int kk0 = kb * 32 + Kb;                    // <= 1592, +7 stays < 1600
  int cin = kk0 / VJ, v = kk0 - cin * VJ;
  int off = cin * 7500 + v;
  int jw  = VJ - v;                          // first j crossing the v=25 wrap

  v8bf pk;
  if (l < LTIME) {
    const float* xp = x + n * (CI * LTIME * VJ) + l * VJ;
    float f[8];
#pragma unroll
    for (int j = 0; j < 8; ++j)
      f[j] = xp[off + j + ((j >= jw) ? 7475 : 0)];
#pragma unroll
    for (int j = 0; j < 8; ++j) pk[j] = (__bf16)f[j];
  } else {
#pragma unroll
    for (int j = 0; j < 8; ++j) pk[j] = (__bf16)0.f;
  }
  *(v8bf*)(xf + (size_t)t * 8) = pk;
}

// ---------------------------------------------------------------------------
// Kernel 3: pure streaming GEMM. No LDS, no barriers. Per wave: 3 A + 2 B
// fragment loads (contiguous 32 B each) + 6 WMMAs per k-step. Accumulator
// rows are consecutive l -> each 16x16 tile stores as two coalesced b128 to
// tpre[n][col][l].
// ---------------------------------------------------------------------------
__global__ __launch_bounds__(256) void gemm_kernel(
    const __bf16* __restrict__ xf, const __bf16* __restrict__ mc,
    float* __restrict__ tpre) {
  const int tid  = threadIdx.x;
  const int lane = tid & 31;
  const int wv   = tid >> 5;
  const int cb   = blockIdx.x;   // 0..6  column block (256 cols)
  const int rg   = blockIdx.y;   // 0..6  row group (3 l-tiles)
  const int n    = blockIdx.z;   // 0..31
  const int lt0  = rg * 3;

  const v16bf* a0p = (const v16bf*)xf + (size_t)((n * LTPAD + lt0 + 0) * KSTEPS) * 32 + lane;
  const v16bf* a1p = (const v16bf*)xf + (size_t)((n * LTPAD + lt0 + 1) * KSTEPS) * 32 + lane;
  const v16bf* a2p = (const v16bf*)xf + (size_t)((n * LTPAD + lt0 + 2) * KSTEPS) * 32 + lane;
  const v16bf* b0p = (const v16bf*)mc + (size_t)(cb * 16 + wv * 2 + 0) * 32 + lane;
  const v16bf* b1p = (const v16bf*)mc + (size_t)(cb * 16 + wv * 2 + 1) * 32 + lane;

  v8f acc00 = {}, acc01 = {}, acc10 = {}, acc11 = {}, acc20 = {}, acc21 = {};

  for (int kb = 0; kb < KSTEPS; ++kb) {
    v16bf a0 = *a0p, a1 = *a1p, a2 = *a2p;
    v16bf b0 = *b0p, b1 = *b1p;
    a0p += 32; a1p += 32; a2p += 32;
    b0p += CTILES * 32; b1p += CTILES * 32;
    __builtin_prefetch(a0p, 0, 0);
    __builtin_prefetch(a1p, 0, 0);
    __builtin_prefetch(a2p, 0, 0);
    __builtin_prefetch(b0p, 0, 0);
    __builtin_prefetch(b1p, 0, 0);

    acc00 = __builtin_amdgcn_wmma_f32_16x16x32_bf16(false, a0, false, b0,
                                                    (short)0, acc00, false, false);
    acc10 = __builtin_amdgcn_wmma_f32_16x16x32_bf16(false, a1, false, b0,
                                                    (short)0, acc10, false, false);
    acc20 = __builtin_amdgcn_wmma_f32_16x16x32_bf16(false, a2, false, b0,
                                                    (short)0, acc20, false, false);
    acc01 = __builtin_amdgcn_wmma_f32_16x16x32_bf16(false, a0, false, b1,
                                                    (short)0, acc01, false, false);
    acc11 = __builtin_amdgcn_wmma_f32_16x16x32_bf16(false, a1, false, b1,
                                                    (short)0, acc11, false, false);
    acc21 = __builtin_amdgcn_wmma_f32_16x16x32_bf16(false, a2, false, b1,
                                                    (short)0, acc21, false, false);
  }

  const int colLo = cb * 256 + wv * 32 + (lane & 15);
  const int mofs  = (lane >> 4) * 8;       // C/D layout: VGPR g -> row g + mofs
  auto store_tile = [&](const v8f& acc, int lt, int col) {
    if (lt < LTILES && col < NCOL) {
      size_t idx = ((size_t)(n * NCOL + col)) * LROWS + lt * 16 + mofs;
      *(v8f*)(tpre + idx) = acc;           // 8 consecutive l -> 2x b128
    }
  };
  store_tile(acc00, lt0 + 0, colLo);
  store_tile(acc01, lt0 + 0, colLo + 16);
  store_tile(acc10, lt0 + 1, colLo);
  store_tile(acc11, lt0 + 1, colLo + 16);
  store_tile(acc20, lt0 + 2, colLo);
  store_tile(acc21, lt0 + 2, colLo + 16);
}

// ---------------------------------------------------------------------------
// Kernel 4: sliding 9-tap temporal window + BN + ReLU + residual + ReLU.
// One thread per (n, col), 75-m chunk; tpre rows are contiguous in l.
// ---------------------------------------------------------------------------
__global__ __launch_bounds__(256) void epilogue_kernel(
    const float* __restrict__ tpre, const float* __restrict__ x,
    const float* __restrict__ gamma, const float* __restrict__ beta,
    const float* __restrict__ mean,  const float* __restrict__ var,
    float* __restrict__ out) {
  int gid = blockIdx.x * 256 + threadIdx.x;    // (n, col)
  if (gid >= NN * NCOL) return;
  int n = gid / NCOL, col = gid - n * NCOL;
  int c = col / VJ,   w   = col - c * VJ;

  float scale = gamma[c] * rsqrtf(var[c] + EPSBN);
  float bias  = beta[c] - mean[c] * scale;

  const float* tp = tpre + (size_t)gid * LROWS;
  const int    ob = ((n * CO + c) * LTIME) * VJ + w;
  const int    m0 = blockIdx.y * 75;

  float tsum = 0.f;
#pragma unroll
  for (int i = 1; i <= 8; ++i) {
    int l = m0 - i;
    if (l >= 0) tsum += tp[l];
  }
  for (int m = m0; m < m0 + 75; ++m) {
    tsum += tp[m];
    if (m >= 9) tsum -= tp[m - 9];
    float xn  = fmaxf(tsum * scale + bias, 0.f);
    float res = x[ob + m * VJ];
    out[ob + m * VJ] = fmaxf(xn + res, 0.f);
  }
}

// ---------------------------------------------------------------------------
extern "C" void kernel_launch(void* const* d_in, const int* in_sizes, int n_in,
                              void* d_out, int out_size, void* d_ws, size_t ws_size,
                              hipStream_t stream) {
  (void)in_sizes; (void)n_in; (void)out_size; (void)ws_size;
  const float* x     = (const float*)d_in[0];
  const float* A     = (const float*)d_in[1];
  const float* W     = (const float*)d_in[2];
  const float* gamma = (const float*)d_in[3];
  const float* beta  = (const float*)d_in[4];
  const float* mean  = (const float*)d_in[5];
  const float* var   = (const float*)d_in[6];
  float* out = (float*)d_out;

  unsigned char* ws = (unsigned char*)d_ws;      // ~98 MB total scratch
  __bf16* mc   = (__bf16*)ws;                    //  5.5 MB B fragments
  __bf16* xf   = (__bf16*)(ws + MC_BYTES);       // 32.8 MB A fragments
  float*  tpre = (float*)(ws + MC_BYTES + XF_BYTES);  // 62.3 MB pre-window GEMM out

  build_mc_kernel<<<(MC_ELEMS + 255) / 256, 256, 0, stream>>>(A, W, mc);

  const int xft = NN * LTPAD * KSTEPS * 32 * 2;  // threads (8 elems each)
  build_xfrag_kernel<<<(xft + 255) / 256, 256, 0, stream>>>(x, xf);

  dim3 ggrid(COLB, 7, NN);                       // 7 col blocks x 7 row groups x 32
  gemm_kernel<<<ggrid, 256, 0, stream>>>(xf, mc, tpre);

  dim3 egrid((NN * NCOL + 255) / 256, 4);        // 51200 cols x 4 m-chunks of 75
  epilogue_kernel<<<egrid, 256, 0, stream>>>(tpre, x, gamma, beta, mean, var, out);
}
